// I2A_MambaDuelingModel_78391743086958
// MI455X (gfx1250) — matrix operand order repaired and checked by reading, not compiled, not run on hardware
//
#include <hip/hip_runtime.h>

// ---------------------------------------------------------------------------
// Model dims (compile-time constants from the reference)
// ---------------------------------------------------------------------------
#define NL    8
#define F_IN  32
#define HID   128
#define T_IN  8
#define T_OUT 32
#define Dm    256      // D = 2*HID
#define DIN   512      // 2*D
#define NST   16
#define DCONV 4
#define DTR   16
#define DFF   682      // int(D*4*2/3)
#define ACT6_LD 684    // DFF padded so lda is a multiple of 4 (16B rows)
#define Bb    4
#define Ll    512
#define Mrows 2048     // B*L
#define NACT  5

// ---------------------------------------------------------------------------
// Types
// ---------------------------------------------------------------------------
typedef __attribute__((ext_vector_type(16))) __bf16        v16bf;
typedef __attribute__((ext_vector_type(8)))  float         v8f;
typedef __attribute__((ext_vector_type(4)))  float         v4f;
typedef __attribute__((ext_vector_type(4)))  unsigned int  v4u;
typedef __attribute__((ext_vector_type(2)))  unsigned int  v2u;
typedef __attribute__((ext_vector_type(4)))  int           v4i;

union FragU { v16bf v; v4u q[2]; };

enum { ACT_NONE = 0, ACT_SOFTPLUS = 1 };

#define AS1 __attribute__((address_space(1)))
#define AS3 __attribute__((address_space(3)))

#if defined(__has_builtin)
#if __has_builtin(__builtin_amdgcn_global_load_async_to_lds_b128)
#define HAVE_ASYNC_LDS 1
#endif
#endif

static __device__ __forceinline__ unsigned short f2bf(float f) {
    unsigned int u = __float_as_uint(f);
    u = (u + 0x7FFFu + ((u >> 16) & 1u)) >> 16;
    return (unsigned short)u;
}
static __device__ __forceinline__ unsigned int pack2bf(float a, float b) {
    return (unsigned int)f2bf(a) | ((unsigned int)f2bf(b) << 16);
}
static __device__ __forceinline__ float siluf(float x) {
    return x / (1.0f + __expf(-x));
}
static __device__ __forceinline__ float softplusf(float x) {
    return (x > 20.0f) ? x : log1pf(__expf(x));
}

// ---------------------------------------------------------------------------
// Weight conversion: fp32 [layers][K][N] row-major -> bf16 [layers][Np][Kp],
// zero padded. Makes all GEMM K/N loop guards unnecessary.
// ---------------------------------------------------------------------------
__global__ void convertT_kernel(const float* __restrict__ W,
                                unsigned short* __restrict__ out,
                                int K, int N, int Kp, int Np, int layers)
{
    size_t total = (size_t)layers * Np * Kp;
    size_t idx = (size_t)blockIdx.x * 256 + threadIdx.x;
    if (idx >= total) return;
    size_t per = (size_t)Np * Kp;
    int l = (int)(idx / per);
    size_t rem = idx - (size_t)l * per;
    int n = (int)(rem / Kp);
    int k = (int)(rem - (size_t)n * Kp);
    float v = (k < K && n < N) ? W[(size_t)l * K * N + (size_t)k * N + n] : 0.0f;
    out[idx] = f2bf(v);
}

__global__ void zero_kernel(float* __restrict__ p, int n)
{
    int i = blockIdx.x * 256 + threadIdx.x;
    if (i < n) p[i] = 0.0f;
}

// ---------------------------------------------------------------------------
// WMMA GEMM: C[M,N] = act(A[M,Kp] * BT^T + bias) (+C if accumulate)
//   A : fp32 row-major, lda multiple of 4, M multiple of 64
//   BT: bf16 [Np][Kp] (transposed weights), Kp mult of 32, Np mult of 128
// Tile: 64(M) x 128(N) per 256-thread block; 8 waves; 4 WMMA / wave / K-step.
// B tiles move global->LDS via async-LDS loads (tracked on ASYNCcnt),
// overlapping with the A-side bf16 conversion.
// ---------------------------------------------------------------------------
#define TILE_M 64
#define TILE_N 128
#define TILE_K 32

__global__ __launch_bounds__(256)
void gemm_wmma_kernel(const float* __restrict__ A, int lda,
                      const unsigned short* __restrict__ BT, int Kp,
                      const float* __restrict__ bias,
                      float* __restrict__ C, int ldc,
                      int M, int N, int act, int accumulate)
{
    __shared__ __align__(16) unsigned int sA32[TILE_M * TILE_K / 2]; // 1024 dw
    __shared__ __align__(16) unsigned int sB32[TILE_N * TILE_K / 2]; // 2048 dw

    const int tid   = threadIdx.x;
    const int wave  = tid >> 5;
    const int lane  = tid & 31;
    const int half  = lane >> 4;
    const int l16   = lane & 15;
    const int tileM = blockIdx.y * TILE_M;
    const int tileN = blockIdx.x * TILE_N;
    const int mi    = wave & 3;          // M-subtile
    const int nb0   = (wave >> 2) * 16;  // N-subtiles nb0 + {0,32,64,96}

    v8f acc[4];
    #pragma unroll
    for (int j = 0; j < 4; ++j) acc[j] = (v8f){};

    const v4u* sA4 = (const v4u*)sA32;
    const v4u* sB4 = (const v4u*)sB32;

    // A staging indices: 64 rows x 8 float4-chunks; 4 threads per row
    const int am = tid >> 2;       // 0..63
    const int aq = tid & 3;        // 0..3

    for (int k0 = 0; k0 < Kp; k0 += TILE_K) {
        // ---- stage BT (pure bf16 copy: async global->LDS when available) ----
#ifdef HAVE_ASYNC_LDS
        #pragma unroll
        for (int r = 0; r < 2; ++r) {
            int idx = tid + r * 256;
            int n = idx >> 2, kq = idx & 3;
            __builtin_amdgcn_global_load_async_to_lds_b128(
                (AS1 v4i*)(BT + (size_t)(tileN + n) * Kp + k0 + kq * 8),
                (AS3 v4i*)&sB32[n * 16 + kq * 4],
                0, 0);
        }
#else
        #pragma unroll
        for (int r = 0; r < 2; ++r) {
            int idx = tid + r * 256;
            int n = idx >> 2, kq = idx & 3;
            v4u w = *(const v4u*)(BT + (size_t)(tileN + n) * Kp + k0 + kq * 8);
            *(v4u*)&sB32[n * 16 + kq * 4] = w;
        }
#endif
        // ---- stage A (fp32 -> packed bf16 pairs) ----
        #pragma unroll
        for (int r = 0; r < 2; ++r) {
            int c = (aq + 4 * r) * 4;                       // float col 0..28
            v4f f = *(const v4f*)(A + (size_t)(tileM + am) * lda + k0 + c);
            v2u p;
            p.x = pack2bf(f.x, f.y);
            p.y = pack2bf(f.z, f.w);
            *(v2u*)&sA32[am * 16 + (c >> 1)] = p;
        }
        if (k0 + TILE_K < Kp)
            __builtin_prefetch(A + (size_t)(tileM + am) * lda + k0 + TILE_K, 0, 1);
#ifdef HAVE_ASYNC_LDS
        asm volatile("s_wait_asynccnt 0" ::: "memory");
#endif
        __syncthreads();

        // ---- fragments: documented 16-bit striping = contiguous b128s ----
        FragU fa;
        {
            int r16 = mi * 16 + l16;
            fa.q[0] = sA4[r16 * 4 + half];
            fa.q[1] = sA4[r16 * 4 + 2 + half];
        }
        #pragma unroll
        for (int j = 0; j < 4; ++j) {
            FragU fb;
            int rB = nb0 + 32 * j + l16;
            fb.q[0] = sB4[rB * 4 + half];
            fb.q[1] = sB4[rB * 4 + 2 + half];
            acc[j] = __builtin_amdgcn_wmma_f32_16x16x32_bf16(
                         false, fa.v, false, fb.v, (short)0, acc[j], false, false);
        }
        __syncthreads();
    }

    // ---- epilogue: m = r + 8*half (always < M); guard n only ----
    #pragma unroll
    for (int r = 0; r < 8; ++r) {
        int m = tileM + mi * 16 + half * 8 + r;
        #pragma unroll
        for (int j = 0; j < 4; ++j) {
            int n = tileN + nb0 + 32 * j + l16;
            if (n < N) {
                float v = acc[j][r];
                if (bias) v += bias[n];
                if (act == ACT_SOFTPLUS) v = softplusf(v);
                if (accumulate) v += C[(size_t)m * ldc + n];
                C[(size_t)m * ldc + n] = v;
            }
        }
    }
}

// ---------------------------------------------------------------------------
// DAIN front-end kernels
// ---------------------------------------------------------------------------
__global__ void colmean_kernel(const float* __restrict__ src, float* __restrict__ mean)
{
    int bf = blockIdx.x;             // b*32+f
    int b = bf >> 5, f = bf & 31;
    __shared__ float red[256];
    float s = 0.0f;
    for (int l = threadIdx.x; l < Ll; l += 256)
        s += src[((size_t)b * Ll + l) * F_IN + f];
    red[threadIdx.x] = s; __syncthreads();
    for (int st = 128; st > 0; st >>= 1) {
        if (threadIdx.x < st) red[threadIdx.x] += red[threadIdx.x + st];
        __syncthreads();
    }
    if (threadIdx.x == 0) mean[bf] = red[0] / (float)Ll;
}

__global__ void dain_small1_kernel(const float* __restrict__ mean,
                                   const float* __restrict__ Wm,
                                   float* __restrict__ avgW)
{
    int t = threadIdx.x;
    if (t >= Bb * F_IN) return;
    int b = t >> 5, f = t & 31;
    float s = 0.0f;
    for (int g = 0; g < F_IN; ++g) s += mean[b * F_IN + g] * Wm[f * F_IN + g];
    avgW[t] = s;
}

__global__ void colstd_kernel(const float* __restrict__ src,
                              const float* __restrict__ avgW,
                              float* __restrict__ stdv)
{
    int bf = blockIdx.x;
    int b = bf >> 5, f = bf & 31;
    __shared__ float red[256];
    float mu = avgW[bf];
    float s = 0.0f;
    for (int l = threadIdx.x; l < Ll; l += 256) {
        float v = src[((size_t)b * Ll + l) * F_IN + f] - mu;
        s += v * v;
    }
    red[threadIdx.x] = s; __syncthreads();
    for (int st = 128; st > 0; st >>= 1) {
        if (threadIdx.x < st) red[threadIdx.x] += red[threadIdx.x + st];
        __syncthreads();
    }
    if (threadIdx.x == 0) stdv[bf] = sqrtf(red[0] / (float)Ll);
}

__global__ void dain_small2_kernel(const float* __restrict__ mean,
                                   const float* __restrict__ avgW,
                                   const float* __restrict__ stdv,
                                   const float* __restrict__ Ws,
                                   const float* __restrict__ Wg,
                                   const float* __restrict__ bg,
                                   float* __restrict__ adstd,
                                   float* __restrict__ gateo)
{
    __shared__ float m2s[Bb * F_IN];
    int t = threadIdx.x;
    if (t >= Bb * F_IN) return;
    int b = t >> 5, f = t & 31;
    float ad = 0.0f;
    for (int g = 0; g < F_IN; ++g) ad += stdv[b * F_IN + g] * Ws[f * F_IN + g];
    if (ad <= 1e-8f) ad = 1.0f;
    adstd[t] = ad;
    m2s[t] = (mean[t] - avgW[t]) / ad;
    __syncthreads();
    float s = bg[f];
    for (int g = 0; g < F_IN; ++g) s += m2s[b * F_IN + g] * Wg[f * F_IN + g];
    gateo[t] = 1.0f / (1.0f + __expf(-s));
}

__global__ void xdain_kernel(const float* __restrict__ src,
                             const float* __restrict__ avgW,
                             const float* __restrict__ adstd,
                             const float* __restrict__ gateo,
                             float* __restrict__ out)
{
    int idx = blockIdx.x * 256 + threadIdx.x;
    if (idx >= Mrows * F_IN) return;
    int row = idx >> 5, f = idx & 31;
    int bf = (row >> 9) * F_IN + f;
    out[idx] = (src[idx] - avgW[bf]) / adstd[bf] * gateo[bf];
}

__global__ void temb_kernel(const float* __restrict__ tau,
                            const float* __restrict__ t_w,
                            const float* __restrict__ t_b,
                            const float* __restrict__ t_w0,
                            const float* __restrict__ t_b0,
                            float* __restrict__ temb)
{
    int idx = blockIdx.x * 256 + threadIdx.x;
    if (idx >= Mrows * T_OUT) return;
    int m = idx >> 5, j = idx & 31;
    const float* tr = tau + (size_t)m * T_IN;
    float s;
    if (j < T_OUT - 1) {
        s = t_b[j];
        for (int k = 0; k < T_IN; ++k) s += tr[k] * t_w[k * (T_OUT - 1) + j];
        s = sinf(s);
    } else {
        s = t_b0[0];
        for (int k = 0; k < T_IN; ++k) s += tr[k] * t_w0[k];
    }
    temb[idx] = s;
}

// ---------------------------------------------------------------------------
// Elementwise / LN kernels
// ---------------------------------------------------------------------------
__global__ void ffn_glu_kernel(const float* __restrict__ w1out, float* __restrict__ out)
{
    int idx = blockIdx.x * 256 + threadIdx.x;
    if (idx >= Mrows * ACT6_LD) return;
    int row = idx / ACT6_LD, j = idx % ACT6_LD;
    float r = 0.0f;
    if (j < DFF) {
        float a = w1out[(size_t)row * (2 * DFF) + j];
        float g = w1out[(size_t)row * (2 * DFF) + DFF + j];
        r = a * (0.5f * g * (1.0f + erff(g * 0.70710678f)));
    }
    out[idx] = r;
}

__global__ void ln_kernel(const float* __restrict__ in,
                          const float* __restrict__ w,
                          const float* __restrict__ b,
                          float* __restrict__ out, int N, int relu)
{
    __shared__ float rs[256], rs2[256];
    int row = blockIdx.x;
    const float* x = in + (size_t)row * N;
    float s = 0.0f, s2 = 0.0f;
    for (int j = threadIdx.x; j < N; j += 256) { float v = x[j]; s += v; s2 += v * v; }
    rs[threadIdx.x] = s; rs2[threadIdx.x] = s2; __syncthreads();
    for (int st = 128; st > 0; st >>= 1) {
        if (threadIdx.x < st) { rs[threadIdx.x] += rs[threadIdx.x + st]; rs2[threadIdx.x] += rs2[threadIdx.x + st]; }
        __syncthreads();
    }
    float mean = rs[0] / (float)N;
    float var  = rs2[0] / (float)N - mean * mean;
    float r    = rsqrtf(var + 1e-5f);
    float* o = out + (size_t)row * N;
    for (int j = threadIdx.x; j < N; j += 256) {
        float v = (x[j] - mean) * r * w[j] + b[j];
        if (relu) v = fmaxf(v, 0.0f);
        o[j] = v;
    }
}

__global__ void conv_silu_kernel(const float* __restrict__ uz,
                                 const float* __restrict__ w,
                                 const float* __restrict__ bias,
                                 float* __restrict__ out)
{
    int idx = blockIdx.x * 256 + threadIdx.x;
    if (idx >= Mrows * DIN) return;
    int row = idx >> 9, d = idx & (DIN - 1);
    int b = row >> 9, t = row & (Ll - 1);
    float acc = bias[d];
    #pragma unroll
    for (int j = 0; j < DCONV; ++j) {
        int tt = t - (DCONV - 1) + j;
        if (tt >= 0) acc += w[d * DCONV + j] * uz[((size_t)(b * Ll + tt)) * (2 * DIN) + d];
    }
    out[idx] = siluf(acc);
}

__global__ void ssm_scan_kernel(const float* __restrict__ uact,
                                const float* __restrict__ dtb,
                                const float* __restrict__ xd,
                                const float* __restrict__ A_log,
                                const float* __restrict__ D_p,
                                float* __restrict__ y)
{
    int idx = blockIdx.x * 256 + threadIdx.x;   // (b, d)
    if (idx >= Bb * DIN) return;
    int b = idx >> 9, d = idx & (DIN - 1);
    float a[NST], h[NST];
    #pragma unroll
    for (int n = 0; n < NST; ++n) { a[n] = -__expf(A_log[d * NST + n]); h[n] = 0.0f; }
    float Dp = D_p[d];
    for (int t = 0; t < Ll; ++t) {
        int row = b * Ll + t;
        float dtv = dtb[(size_t)row * DIN + d];
        float uv  = uact[(size_t)row * DIN + d];
        float du  = dtv * uv;
        const float* xr = xd + (size_t)row * (DTR + 2 * NST);
        float yv = 0.0f;
        #pragma unroll
        for (int n = 0; n < NST; ++n) {
            h[n] = __expf(dtv * a[n]) * h[n] + du * xr[DTR + n];
            yv  += h[n] * xr[DTR + NST + n];
        }
        y[(size_t)row * DIN + d] = yv + uv * Dp;
    }
}

__global__ void yz_kernel(const float* __restrict__ y,
                          const float* __restrict__ uz,
                          float* __restrict__ out)
{
    int idx = blockIdx.x * 256 + threadIdx.x;
    if (idx >= Mrows * DIN) return;
    int row = idx >> 9, d = idx & (DIN - 1);
    float z = uz[(size_t)row * (2 * DIN) + DIN + d];
    out[idx] = y[idx] * siluf(z);
}

__global__ void silu_mul_kernel(const float* __restrict__ m1out, float* __restrict__ out)
{
    int idx = blockIdx.x * 256 + threadIdx.x;
    if (idx >= Mrows * 256) return;
    int row = idx >> 8, j = idx & 255;
    float g1 = m1out[(size_t)row * 512 + j];
    float g2 = m1out[(size_t)row * 512 + 256 + j];
    out[idx] = siluf(g1) * g2;
}

__global__ void gather_last_kernel(const float* __restrict__ h, float* __restrict__ rows4)
{
    int idx = blockIdx.x * 256 + threadIdx.x;
    if (idx >= Bb * Dm) return;
    int b = idx >> 8, j = idx & 255;
    rows4[idx] = h[((size_t)(b * Ll + (Ll - 1))) * Dm + j];
}

__global__ void fc_kernel(const float* __restrict__ in,
                          const float* __restrict__ W,
                          const float* __restrict__ bias,
                          float* __restrict__ out, int M, int N, int K)
{
    int idx = blockIdx.x * 256 + threadIdx.x;
    if (idx >= M * N) return;
    int m = idx / N, n = idx % N;
    float s = bias ? bias[n] : 0.0f;
    for (int k = 0; k < K; ++k) s += in[(size_t)m * K + k] * W[(size_t)k * N + n];
    out[idx] = s;
}

__global__ void combine_kernel(const float* __restrict__ val,
                               const float* __restrict__ adv,
                               float* __restrict__ q)
{
    int t = threadIdx.x;
    if (t >= Bb * NACT) return;
    int b = t / NACT, a = t % NACT;
    float m = 0.0f;
    for (int j = 0; j < NACT; ++j) m += adv[b * NACT + j];
    m *= (1.0f / NACT);
    q[t] = val[b] + adv[b * NACT + a] - m;
}

// ---------------------------------------------------------------------------
// Host orchestration
// ---------------------------------------------------------------------------
static void launch_gemm(hipStream_t s, const float* A, int lda,
                        const unsigned short* BT, int Kp,
                        const float* bias, float* C, int ldc,
                        int M, int N, int act, int accum)
{
    dim3 grid((N + TILE_N - 1) / TILE_N, M / TILE_M);
    gemm_wmma_kernel<<<grid, 256, 0, s>>>(A, lda, BT, Kp, bias, C, ldc, M, N, act, accum);
}

extern "C" void kernel_launch(void* const* d_in, const int* in_sizes, int n_in,
                              void* d_out, int out_size, void* d_ws, size_t ws_size,
                              hipStream_t stream)
{
    (void)in_sizes; (void)n_in; (void)out_size; (void)ws_size;

    const float* src       = (const float*)d_in[0];
    const float* time_tau  = (const float*)d_in[1];
    const float* dain_Wm   = (const float*)d_in[2];
    const float* dain_Ws   = (const float*)d_in[3];
    const float* dain_Wg   = (const float*)d_in[4];
    const float* dain_bg   = (const float*)d_in[5];
    const float* memb_W    = (const float*)d_in[6];
    const float* memb_b    = (const float*)d_in[7];
    const float* t_w0      = (const float*)d_in[8];
    const float* t_b0      = (const float*)d_in[9];
    const float* t_w       = (const float*)d_in[10];
    const float* t_b       = (const float*)d_in[11];
    const float* tproj_W   = (const float*)d_in[12];
    const float* tproj_b   = (const float*)d_in[13];
    const float* ffn_W1    = (const float*)d_in[14];
    const float* ffn_b1    = (const float*)d_in[15];
    const float* ffn_W2    = (const float*)d_in[16];
    const float* ffn_b2    = (const float*)d_in[17];
    const float* n1_w      = (const float*)d_in[18];
    const float* n1_b      = (const float*)d_in[19];
    const float* inproj_W  = (const float*)d_in[20];
    const float* conv_w    = (const float*)d_in[21];
    const float* conv_b    = (const float*)d_in[22];
    const float* xproj_W   = (const float*)d_in[23];
    const float* dtproj_W  = (const float*)d_in[24];
    const float* dt_bias   = (const float*)d_in[25];
    const float* A_log     = (const float*)d_in[26];
    const float* D_p       = (const float*)d_in[27];
    const float* outproj_W = (const float*)d_in[28];
    const float* n2_w      = (const float*)d_in[29];
    const float* n2_b      = (const float*)d_in[30];
    const float* mlp_W1    = (const float*)d_in[31];
    const float* mlp_W2    = (const float*)d_in[32];
    const float* nf_w      = (const float*)d_in[33];
    const float* nf_b      = (const float*)d_in[34];
    const float* val_W1    = (const float*)d_in[35];
    const float* val_b1    = (const float*)d_in[36];
    const float* val_ln1w  = (const float*)d_in[37];
    const float* val_ln1b  = (const float*)d_in[38];
    const float* val_W2    = (const float*)d_in[39];
    const float* val_b2    = (const float*)d_in[40];
    const float* val_ln2w  = (const float*)d_in[41];
    const float* val_ln2b  = (const float*)d_in[42];
    const float* val_W3    = (const float*)d_in[43];
    const float* val_b3    = (const float*)d_in[44];
    const float* adv_W1    = (const float*)d_in[45];
    const float* adv_b1    = (const float*)d_in[46];
    const float* adv_ln1w  = (const float*)d_in[47];
    const float* adv_ln1b  = (const float*)d_in[48];
    const float* adv_W2    = (const float*)d_in[49];
    const float* adv_b2    = (const float*)d_in[50];
    const float* adv_ln2w  = (const float*)d_in[51];
    const float* adv_ln2b  = (const float*)d_in[52];
    const float* adv_W3    = (const float*)d_in[53];
    const float* adv_b3    = (const float*)d_in[54];

    float* ws = (float*)d_ws;
    // ---- fp32 activation workspace ----
    float* mean   = ws + 0;
    float* avgW   = ws + 128;
    float* stdv   = ws + 256;
    float* adstd  = ws + 384;
    float* gateo  = ws + 512;
    size_t off = 1024;
    float* xdain  = ws + off; off += (size_t)Mrows * F_IN;
    float* temb   = ws + off; off += (size_t)Mrows * T_OUT;
    float* h      = ws + off; off += (size_t)Mrows * Dm;
    float* hn     = ws + off; off += (size_t)Mrows * Dm;
    float* big    = ws + off; off += (size_t)Mrows * (2 * DFF); // ffn1out / uz(1024) / mlp1out(512)
    float* act6   = ws + off; off += (size_t)Mrows * ACT6_LD + 64; // +64 slack (zeroed)
    float* uact   = ws + off; off += (size_t)Mrows * DIN;
    float* xd     = ws + off; off += (size_t)Mrows * (DTR + 2 * NST);
    float* dtb    = ws + off; off += (size_t)Mrows * DIN;
    float* ybuf   = ws + off; off += (size_t)Mrows * DIN;
    float* yz     = ws + off; off += (size_t)Mrows * DIN;
    float* mlpact = ws + off; off += (size_t)Mrows * 256;
    float* rows4  = ws + off; off += (size_t)Bb * Dm;
    float* fbuf   = ws + off; off += (size_t)Bb * Dm;
    float* t1     = ws + off; off += (size_t)Bb * 512;
    float* t1l    = ws + off; off += (size_t)Bb * 512;
    float* t2     = ws + off; off += (size_t)Bb * 256;
    float* t2l    = ws + off; off += (size_t)Bb * 256;
    float* valo   = ws + off; off += 16;
    float* advo   = ws + off; off += 32;
    off = (off + 255) & ~(size_t)255;      // align bf16 pool to 1KB

    // ---- bf16 transposed-weight pool ----
    unsigned short* wsT = (unsigned short*)(ws + off);
    size_t offT = 0;
    unsigned short* membT   = wsT + offT; offT += (size_t)128 * 32;
    unsigned short* tprojT  = wsT + offT; offT += (size_t)128 * 32;
    unsigned short* ffn1T   = wsT + offT; offT += (size_t)1408 * 256;
    unsigned short* ffn2T   = wsT + offT; offT += (size_t)256 * 704;
    unsigned short* inprojT = wsT + offT; offT += (size_t)NL * 1024 * 256;
    unsigned short* xprojT  = wsT + offT; offT += (size_t)NL * 128 * 512;
    unsigned short* dtprojT = wsT + offT; offT += (size_t)NL * 512 * 32;
    unsigned short* outprjT = wsT + offT; offT += (size_t)NL * 256 * 512;
    unsigned short* mlp1T   = wsT + offT; offT += (size_t)NL * 512 * 256;
    unsigned short* mlp2T   = wsT + offT; offT += (size_t)NL * 256 * 256;

    const dim3 blk(256);
    auto convT = [&](const float* W, unsigned short* o, int K, int N, int Kp, int Np, int layers) {
        size_t total = (size_t)layers * Np * Kp;
        convertT_kernel<<<(unsigned)((total + 255) / 256), blk, 0, stream>>>(W, o, K, N, Kp, Np, layers);
    };

    // ---- weight conversion (every call; deterministic) ----
    convT(memb_W,    membT,   F_IN, HID,          32,  128,  1);
    convT(tproj_W,   tprojT,  T_OUT, HID,         32,  128,  1);
    convT(ffn_W1,    ffn1T,   Dm,   2 * DFF,      256, 1408, 1);
    convT(ffn_W2,    ffn2T,   DFF,  Dm,           704, 256,  1);
    convT(inproj_W,  inprojT, Dm,   2 * DIN,      256, 1024, NL);
    convT(xproj_W,   xprojT,  DIN,  DTR + 2*NST,  512, 128,  NL);
    convT(dtproj_W,  dtprojT, DTR,  DIN,          32,  512,  NL);
    convT(outproj_W, outprjT, DIN,  Dm,           512, 256,  NL);
    convT(mlp_W1,    mlp1T,   Dm,   512,          256, 512,  NL);
    convT(mlp_W2,    mlp2T,   256,  Dm,           256, 256,  NL);
    zero_kernel<<<1, 256, 0, stream>>>(act6 + (size_t)Mrows * ACT6_LD, 64);

    // ---- DAIN front-end ----
    colmean_kernel<<<Bb * F_IN, blk, 0, stream>>>(src, mean);
    dain_small1_kernel<<<1, 128, 0, stream>>>(mean, dain_Wm, avgW);
    colstd_kernel<<<Bb * F_IN, blk, 0, stream>>>(src, avgW, stdv);
    dain_small2_kernel<<<1, 128, 0, stream>>>(mean, avgW, stdv, dain_Ws, dain_Wg, dain_bg, adstd, gateo);
    xdain_kernel<<<(Mrows * F_IN + 255) / 256, blk, 0, stream>>>(src, avgW, adstd, gateo, xdain);
    temb_kernel<<<(Mrows * T_OUT + 255) / 256, blk, 0, stream>>>(time_tau, t_w, t_b, t_w0, t_b0, temb);

    // market -> h[:, :128], tproj -> h[:, 128:]
    launch_gemm(stream, xdain, F_IN, membT, 32, memb_b, h, Dm, Mrows, HID, ACT_NONE, 0);
    launch_gemm(stream, temb, T_OUT, tprojT, 32, tproj_b, h + HID, Dm, Mrows, HID, ACT_NONE, 0);

    // ---- front FFN (gated GELU) ----
    launch_gemm(stream, h, Dm, ffn1T, 256, ffn_b1, big, 2 * DFF, Mrows, 2 * DFF, ACT_NONE, 0);
    ffn_glu_kernel<<<(Mrows * ACT6_LD + 255) / 256, blk, 0, stream>>>(big, act6);
    launch_gemm(stream, act6, ACT6_LD, ffn2T, 704, ffn_b2, h, Dm, Mrows, Dm, ACT_NONE, 0);

    // ---- Mamba layers ----
    for (int i = 0; i < NL; ++i) {
        ln_kernel<<<Mrows, blk, 0, stream>>>(h, n1_w + i * Dm, n1_b + i * Dm, hn, Dm, 0);
        launch_gemm(stream, hn, Dm, inprojT + (size_t)i * 1024 * 256, 256,
                    nullptr, big, 2 * DIN, Mrows, 2 * DIN, ACT_NONE, 0);
        conv_silu_kernel<<<(Mrows * DIN + 255) / 256, blk, 0, stream>>>(
            big, conv_w + (size_t)i * DIN * DCONV, conv_b + i * DIN, uact);
        launch_gemm(stream, uact, DIN, xprojT + (size_t)i * 128 * 512, 512,
                    nullptr, xd, DTR + 2 * NST, Mrows, DTR + 2 * NST, ACT_NONE, 0);
        launch_gemm(stream, xd, DTR + 2 * NST, dtprojT + (size_t)i * 512 * 32, 32,
                    dt_bias + i * DIN, dtb, DIN, Mrows, DIN, ACT_SOFTPLUS, 0);
        ssm_scan_kernel<<<(Bb * DIN + 255) / 256, blk, 0, stream>>>(
            uact, dtb, xd, A_log + (size_t)i * DIN * NST, D_p + i * DIN, ybuf);
        yz_kernel<<<(Mrows * DIN + 255) / 256, blk, 0, stream>>>(ybuf, big, yz);
        launch_gemm(stream, yz, DIN, outprjT + (size_t)i * 256 * 512, 512,
                    nullptr, h, Dm, Mrows, Dm, ACT_NONE, 1);
        ln_kernel<<<Mrows, blk, 0, stream>>>(h, n2_w + i * Dm, n2_b + i * Dm, hn, Dm, 0);
        launch_gemm(stream, hn, Dm, mlp1T + (size_t)i * 512 * 256, 256,
                    nullptr, big, 512, Mrows, 512, ACT_NONE, 0);
        silu_mul_kernel<<<(Mrows * 256 + 255) / 256, blk, 0, stream>>>(big, mlpact);
        launch_gemm(stream, mlpact, 256, mlp2T + (size_t)i * 256 * 256, 256,
                    nullptr, h, Dm, Mrows, Dm, ACT_NONE, 1);
    }

    // ---- final LN + dueling heads ----
    gather_last_kernel<<<(Bb * Dm + 255) / 256, blk, 0, stream>>>(h, rows4);
    ln_kernel<<<Bb, blk, 0, stream>>>(rows4, nf_w, nf_b, fbuf, Dm, 0);

    // value head
    fc_kernel<<<(Bb * 512 + 255) / 256, blk, 0, stream>>>(fbuf, val_W1, val_b1, t1, Bb, 512, Dm);
    ln_kernel<<<Bb, blk, 0, stream>>>(t1, val_ln1w, val_ln1b, t1l, 512, 1);
    fc_kernel<<<(Bb * 256 + 255) / 256, blk, 0, stream>>>(t1l, val_W2, val_b2, t2, Bb, 256, 512);
    ln_kernel<<<Bb, blk, 0, stream>>>(t2, val_ln2w, val_ln2b, t2l, 256, 1);
    fc_kernel<<<1, blk, 0, stream>>>(t2l, val_W3, val_b3, valo, Bb, 1, 256);

    // advantage head
    fc_kernel<<<(Bb * 512 + 255) / 256, blk, 0, stream>>>(fbuf, adv_W1, adv_b1, t1, Bb, 512, Dm);
    ln_kernel<<<Bb, blk, 0, stream>>>(t1, adv_ln1w, adv_ln1b, t1l, 512, 1);
    fc_kernel<<<(Bb * 256 + 255) / 256, blk, 0, stream>>>(t1l, adv_W2, adv_b2, t2, Bb, 256, 512);
    ln_kernel<<<Bb, blk, 0, stream>>>(t2, adv_ln2w, adv_ln2b, t2l, 256, 1);
    fc_kernel<<<1, blk, 0, stream>>>(t2l, adv_W3, adv_b3, advo, Bb, NACT, 256);

    combine_kernel<<<1, 32, 0, stream>>>(valo, advo, (float*)d_out);
}
